// EdgeGAT_16647293239655
// MI455X (gfx1250) — matrix operand order repaired and checked by reading, not compiled, not run on hardware
//
#include <hip/hip_runtime.h>
#include <hip/hip_bf16.h>
#include <math.h>

#define NODES 100000
#define EDGES 3200000
#define INCH  128
#define HID   64
#define NLAYERS 3
#define NGRAPH  64
#define NCLS    5
#define NEG_SLOPE 0.2f

typedef __attribute__((ext_vector_type(16))) _Float16 v16h;
typedef __attribute__((ext_vector_type(8)))  float    v8f;

// ---------------------------------------------------------------------------
// One-shot: Wt[n][k] = (f16) W[k][n]  (transposed f16 copy of K x 64 weight)
// ---------------------------------------------------------------------------
__global__ void convert_w_t_f16(const float* __restrict__ W,
                                _Float16* __restrict__ Wt, int K) {
  int i = blockIdx.x * blockDim.x + threadIdx.x;
  if (i >= HID * K) return;
  int n = i / K, k = i - n * K;
  Wt[i] = (_Float16)W[(size_t)k * HID + n];
}

// ---------------------------------------------------------------------------
// C[NODES x 64] = A[NODES x K] * B[K x 64]; A fp32 row-major, B given as
// transposed f16 (Bt[64][K]).  f16 WMMA, f32 accumulate.
// Block = 256 threads = 8 waves over a 128-row x 64-col super-tile:
// wave w -> col tile (w&3), row tiles (w>>2)*64 + t*16, t=0..3.
// Tail blocks clamp the row tile to the last valid tile (NODES % 16 == 0),
// recomputing/restoring identical values -- branch-free, EXEC stays full.
// ---------------------------------------------------------------------------
template <int K>
__global__ void __launch_bounds__(256)
gat_gemm_wmma(const float* __restrict__ A, const _Float16* __restrict__ Bt,
              float* __restrict__ C) {
  const int wave  = threadIdx.x >> 5;
  const int lane  = threadIdx.x & 31;
  const int col0  = (wave & 3) << 4;
  const int rbase = blockIdx.x * 128 + (wave >> 2) * 64;
  const int m     = lane & 15;     // A row in tile / B,D column
  const int hi    = lane >> 4;     // lane-half selector

  int rtile[4];
#pragma unroll
  for (int t = 0; t < 4; ++t) rtile[t] = min(rbase + t * 16, NODES - 16);

  v8f acc[4] = {};
  const _Float16* bt = Bt + (size_t)(col0 + m) * K + hi * 16;

#pragma unroll
  for (int kb = 0; kb < K; kb += 32) {
    // B fragment: halves j -> K = kb + hi*16 + j, contiguous in Bt -> 32B load
    v16h b = *(const v16h*)(bt + kb);
#pragma unroll
    for (int t = 0; t < 4; ++t) {
      // A fragment: lanes 0-15 K = kb+{0..15}; lanes 16-31 K = kb+{16..31}
      const float4* a4 = (const float4*)(A + (size_t)(rtile[t] + m) * K + kb + hi * 16);
      float4 p0 = a4[0], p1 = a4[1], p2 = a4[2], p3 = a4[3];
      v16h a;
      a[0]  = (_Float16)p0.x; a[1]  = (_Float16)p0.y;
      a[2]  = (_Float16)p0.z; a[3]  = (_Float16)p0.w;
      a[4]  = (_Float16)p1.x; a[5]  = (_Float16)p1.y;
      a[6]  = (_Float16)p1.z; a[7]  = (_Float16)p1.w;
      a[8]  = (_Float16)p2.x; a[9]  = (_Float16)p2.y;
      a[10] = (_Float16)p2.z; a[11] = (_Float16)p2.w;
      a[12] = (_Float16)p3.x; a[13] = (_Float16)p3.y;
      a[14] = (_Float16)p3.z; a[15] = (_Float16)p3.w;
      acc[t] = __builtin_amdgcn_wmma_f32_16x16x32_f16(false, a, false, b,
                                                      (short)0, acc[t], false, false);
    }
  }
  // D layout: VGPR r, lanes 0-15 -> row r, lanes 16-31 -> row r+8; col = m
#pragma unroll
  for (int t = 0; t < 4; ++t) {
    float* crow = C + (size_t)(rtile[t] + hi * 8) * HID + col0 + m;
#pragma unroll
    for (int r = 0; r < 8; ++r) crow[(size_t)r * HID] = acc[t][r];
  }
}

// ---------------------------------------------------------------------------
// Per-node attention scores s_src = h.a_src, s_dst = h.a_dst; init segment
// max/-inf, denom/0 and zero the aggregation buffer.
// ---------------------------------------------------------------------------
__global__ void gat_node_prep(const float* __restrict__ h,
                              const float* __restrict__ a_src,
                              const float* __restrict__ a_dst,
                              float* __restrict__ s_src, float* __restrict__ s_dst,
                              float* __restrict__ segmax, float* __restrict__ denom,
                              float* __restrict__ out_accum) {
  int n = blockIdx.x * blockDim.x + threadIdx.x;
  if (n >= NODES) return;
  const float4* hr = (const float4*)(h + (size_t)n * HID);
  const float4* as = (const float4*)a_src;
  const float4* ad = (const float4*)a_dst;
  float4* oz = (float4*)(out_accum + (size_t)n * HID);
  float ss = 0.f, sd = 0.f;
#pragma unroll
  for (int j = 0; j < HID / 4; ++j) {
    float4 v = hr[j], s = as[j], d = ad[j];
    ss += v.x * s.x + v.y * s.y + v.z * s.z + v.w * s.w;
    sd += v.x * d.x + v.y * d.y + v.z * d.z + v.w * d.w;
    oz[j] = make_float4(0.f, 0.f, 0.f, 0.f);
  }
  s_src[n] = ss;
  s_dst[n] = sd;
  segmax[n] = -INFINITY;
  denom[n] = 0.f;
}

__device__ inline float atomicMaxF(float* addr, float val) {
  if (val >= 0.f)
    return __int_as_float(atomicMax((int*)addr, __float_as_int(val)));
  return __uint_as_float(atomicMin((unsigned int*)addr, __float_as_uint(val)));
}

// logit = leaky_relu(s_src[src] + s_dst[dst]); segment-max over dst
__global__ void gat_edge_logit(const int* __restrict__ ei,
                               const float* __restrict__ s_src,
                               const float* __restrict__ s_dst,
                               float* __restrict__ ew,
                               float* __restrict__ segmax, int etot) {
  int e = blockIdx.x * blockDim.x + threadIdx.x;
  if (e >= etot) return;
  int s, d;
  if (e < EDGES) { s = ei[e]; d = ei[EDGES + e]; } else { s = d = e - EDGES; }
  float l = s_src[s] + s_dst[d];
  l = l >= 0.f ? l : NEG_SLOPE * l;
  ew[e] = l;
  atomicMaxF(&segmax[d], l);
}

// w = exp(logit - max[dst]); segment-sum of w over dst
__global__ void gat_edge_exp(const int* __restrict__ ei,
                             float* __restrict__ ew,
                             const float* __restrict__ segmax,
                             float* __restrict__ denom, int etot) {
  int e = blockIdx.x * blockDim.x + threadIdx.x;
  if (e >= etot) return;
  int d = (e < EDGES) ? ei[EDGES + e] : e - EDGES;
  float mm = segmax[d];
  if (!(mm > -1e38f && mm < 1e38f)) mm = 0.f;  // reference's isfinite guard
  float w = __expf(ew[e] - mm);
  ew[e] = w;
  atomicAdd(&denom[d], w);
}

// One wave32 per edge: alpha = w/denom[dst]; out[dst] += alpha * h[src]
__global__ void __launch_bounds__(256)
gat_edge_scatter(const int* __restrict__ ei, const float* __restrict__ ew,
                 const float* __restrict__ denom, const float* __restrict__ h,
                 float* __restrict__ out, int etot) {
  int e = (blockIdx.x * blockDim.x + threadIdx.x) >> 5;
  if (e >= etot) return;
  int lane = threadIdx.x & 31;
  int s, d;
  if (e < EDGES) { s = ei[e]; d = ei[EDGES + e]; } else { s = d = e - EDGES; }
  float alpha = ew[e] / denom[d];
  float2 v = ((const float2*)(h + (size_t)s * HID))[lane];
  float* ob = out + (size_t)d * HID + lane * 2;
  atomicAdd(ob,     alpha * v.x);
  atomicAdd(ob + 1, alpha * v.y);
}

__global__ void gat_bias_act(float* __restrict__ h, const float* __restrict__ bias,
                             int apply_elu) {
  int i = blockIdx.x * blockDim.x + threadIdx.x;
  if (i >= NODES * HID) return;
  float v = h[i] + bias[i & (HID - 1)];
  if (apply_elu) v = v > 0.f ? v : __expf(v) - 1.f;
  h[i] = v;
}

__global__ void pool_init(float* __restrict__ psum, float* __restrict__ pcnt) {
  int i = blockIdx.x * blockDim.x + threadIdx.x;
  if (i < NGRAPH * HID) psum[i] = 0.f;
  if (i < NGRAPH) pcnt[i] = 0.f;
}

__global__ void pool_scatter(const float* __restrict__ h, const int* __restrict__ batch,
                             float* __restrict__ psum, float* __restrict__ pcnt) {
  int i = blockIdx.x * blockDim.x + threadIdx.x;
  if (i >= NODES * HID) return;
  int n = i >> 6, c = i & 63;
  int g = batch[n];
  atomicAdd(&psum[g * HID + c], h[i]);
  if (c == 0) atomicAdd(&pcnt[g], 1.f);
}

// One block of 64 threads; thread g computes graph g's MLP head + softmax.
__global__ void __launch_bounds__(64)
gat_head(const float* __restrict__ psum, const float* __restrict__ pcnt,
         const float* __restrict__ sex, const float* __restrict__ cag,
         const float* __restrict__ l1w, const float* __restrict__ l1b,
         const float* __restrict__ l2w, const float* __restrict__ l2b,
         float* __restrict__ out) {
  __shared__ float pooled[NGRAPH * HID];
  int t = threadIdx.x;
  for (int i = t; i < NGRAPH * HID; i += 64) {
    int g = i >> 6;
    pooled[i] = psum[i] / fmaxf(pcnt[g], 1.f);
  }
  __syncthreads();
  int g = t;
  float x1[HID];
  for (int c = 0; c < HID; ++c) {
    float acc = l1b[c];
    for (int k = 0; k < HID; ++k) acc += pooled[g * HID + k] * l1w[k * HID + c];
    x1[c] = fmaxf(acc, 0.f);
  }
  float logits[NCLS];
  float mx = -INFINITY;
  for (int j = 0; j < NCLS; ++j) {
    float acc = l2b[j];
    for (int c = 0; c < HID; ++c) acc += x1[c] * l2w[c * NCLS + j];
    acc += sex[g] * l2w[HID * NCLS + j];
    acc += cag[g] * l2w[(HID + 1) * NCLS + j];
    logits[j] = acc;
    mx = fmaxf(mx, acc);
  }
  float se = 0.f;
  for (int j = 0; j < NCLS; ++j) { logits[j] = __expf(logits[j] - mx); se += logits[j]; }
  for (int j = 0; j < NCLS; ++j) out[g * NCLS + j] = logits[j] / se;
}

extern "C" void kernel_launch(void* const* d_in, const int* in_sizes, int n_in,
                              void* d_out, int out_size, void* d_ws, size_t ws_size,
                              hipStream_t stream) {
  (void)in_sizes; (void)n_in; (void)out_size; (void)ws_size;
  const float* x     = (const float*)d_in[0];
  const int*   ei    = (const int*)d_in[1];
  const int*   batch = (const int*)d_in[2];
  const float* sex   = (const float*)d_in[3];
  const float* cag   = (const float*)d_in[4];
  const float* W0    = (const float*)d_in[5];
  const float* Ws    = (const float*)d_in[6];
  const float* asrc  = (const float*)d_in[7];
  const float* adst  = (const float*)d_in[8];
  const float* cbias = (const float*)d_in[9];
  const float* l1w   = (const float*)d_in[10];
  const float* l1b   = (const float*)d_in[11];
  const float* l2w   = (const float*)d_in[12];
  const float* l2b   = (const float*)d_in[13];
  float* out = (float*)d_out;

  const int ETOT = EDGES + NODES;  // with self-loops

  float* ws = (float*)d_ws;
  float* hA    = ws; ws += (size_t)NODES * HID;   // W x (WMMA output)
  float* hB    = ws; ws += (size_t)NODES * HID;   // aggregation target
  float* ssrc  = ws; ws += NODES;
  float* sdst  = ws; ws += NODES;
  float* smax  = ws; ws += NODES;
  float* denom = ws; ws += NODES;
  float* ew    = ws; ws += ETOT;
  float* psum  = ws; ws += NGRAPH * HID;
  float* pcnt  = ws; ws += NGRAPH;
  _Float16* wt16 = (_Float16*)ws;                 // transposed f16 weight (<=16KB)

  const float* hin = x;
  const int gemmGrid = (NODES + 127) / 128;
  for (int layer = 0; layer < NLAYERS; ++layer) {
    const int K = (layer == 0) ? INCH : HID;
    const float* W = (layer == 0) ? W0 : Ws + (size_t)(layer - 1) * HID * HID;

    convert_w_t_f16<<<(HID * K + 255) / 256, 256, 0, stream>>>(W, wt16, K);
    if (K == INCH)
      gat_gemm_wmma<INCH><<<gemmGrid, 256, 0, stream>>>(hin, wt16, hA);
    else
      gat_gemm_wmma<HID><<<gemmGrid, 256, 0, stream>>>(hin, wt16, hA);

    gat_node_prep<<<(NODES + 255) / 256, 256, 0, stream>>>(
        hA, asrc + layer * HID, adst + layer * HID, ssrc, sdst, smax, denom, hB);
    gat_edge_logit<<<(ETOT + 255) / 256, 256, 0, stream>>>(ei, ssrc, sdst, ew, smax, ETOT);
    gat_edge_exp<<<(ETOT + 255) / 256, 256, 0, stream>>>(ei, ew, smax, denom, ETOT);
    gat_edge_scatter<<<(ETOT + 7) / 8, 256, 0, stream>>>(ei, ew, denom, hA, hB, ETOT);
    gat_bias_act<<<((size_t)NODES * HID + 255) / 256, 256, 0, stream>>>(
        hB, cbias + layer * HID, layer < NLAYERS - 1);
    hin = hB;
  }

  pool_init<<<(NGRAPH * HID + 255) / 256, 256, 0, stream>>>(psum, pcnt);
  pool_scatter<<<((size_t)NODES * HID + 255) / 256, 256, 0, stream>>>(hB, batch, psum, pcnt);
  gat_head<<<1, 64, 0, stream>>>(psum, pcnt, sex, cag, l1w, l1b, l2w, l2b, out);
}
